// SSMBlock_19318762897458
// MI455X (gfx1250) — compile-verified, matrix operand order checked by reading, not run on hardware
//
#include <hip/hip_runtime.h>
#include <hip/hip_bf16.h>
#include <stdint.h>

typedef __attribute__((ext_vector_type(16))) _Float16 v16h;
typedef __attribute__((ext_vector_type(8)))  _Float16 v8h;
typedef __attribute__((ext_vector_type(8)))  float    v8f;

union FragU { v16h v; v8h h[2]; };

#define BSZ    8
#define DIM    128
#define NSEQ   2048
#define DSTATE 128
#define TROWS  32        // time-rows per workgroup tile (GEMM kernels)
#define LPAD   136       // f16 LDS row stride in halfs (272 B: conflict-free, 16B aligned)
#define CHUNK  128       // scan chunk length
#define NCH    (NSEQ / CHUNK)   // 16 chunks per batch

#define WMMA_F16(A, B, C) \
    __builtin_amdgcn_wmma_f32_16x16x32_f16(false, (A), false, (B), (short)0, (C), false, false)

// ---------------------------------------------------------------------------
// Fragment loaders.
// A-matrix (16-bit, 16x32, ISA 7.12.2): lane L holds row m=L&15; half=L>>4.
// v16h elems [0..7]  = local K 8*half .. 8*half+7
// v16h elems [8..15] = local K 16+8*half .. +7
// For a row-major f16 LDS tile these are two contiguous 16B runs.
// ---------------------------------------------------------------------------
__device__ __forceinline__ v16h load_a_frag(const _Float16* base, int m0, int ck, int lane) {
    int m = m0 + (lane & 15);
    int hf = lane >> 4;
    const _Float16* p = base + m * LPAD + ck * 32 + hf * 8;
    FragU f;
    f.h[0] = *(const v8h*)(p);
    f.h[1] = *(const v8h*)(p + 16);
    return f.v;
}

// B-matrix fragments pre-swizzled by prep_weights into per-lane order:
// slot = ((ntile*4 + ck)*32 + lane), 16 halfs per slot (32 B, two b128 loads).
__device__ __forceinline__ v16h load_b_frag(const _Float16* wbase, int ntile, int ck, int lane) {
    const _Float16* p = wbase + ((((ntile << 2) + ck) << 5) + lane) * 16;
    FragU f;
    f.h[0] = *(const v8h*)(p);
    f.h[1] = *(const v8h*)(p + 8);
    return f.v;
}

__device__ __forceinline__ float sigmoidf_fast(float x) {
    return 1.0f / (1.0f + __expf(-x));
}

// ---------------------------------------------------------------------------
// Swizzle the four 128x128 weight matrices (all [out][in] row-major; GEMM uses
// out as N and in as K) into WMMA-B fragment order, f32 -> f16.
// ---------------------------------------------------------------------------
__global__ __launch_bounds__(256) void prep_weights(const float* __restrict__ Win,
                                                    const float* __restrict__ Bw,
                                                    const float* __restrict__ Cw,
                                                    const float* __restrict__ Wout,
                                                    _Float16* __restrict__ w16) {
    int gid  = blockIdx.x * 256 + threadIdx.x;      // 0..4095
    int mat  = gid >> 10;
    int rem  = gid & 1023;
    int ntile = rem >> 7;
    int ck    = (rem >> 5) & 3;
    int lane  = rem & 31;
    const float* src = (mat == 0) ? Win : (mat == 1) ? Bw : (mat == 2) ? Cw : Wout;
    _Float16* dst = w16 + mat * (128 * 128) + ((((ntile << 2) + ck) << 5) + lane) * 16;
    int n  = (ntile << 4) + (lane & 15);
    int hf = lane >> 4;
    int kb = ck * 32;
    #pragma unroll
    for (int e = 0; e < 8; ++e) dst[e]     = (_Float16)src[n * 128 + kb + 8 * hf + e];
    #pragma unroll
    for (int e = 0; e < 8; ++e) dst[8 + e] = (_Float16)src[n * 128 + kb + 16 + 8 * hf + e];
}

// ---------------------------------------------------------------------------
// Front end: conv+SiLU -> xp = xc*W_in^T + b_in -> Bbar = (xp*B_w^T)*sigma(xp),
// Cm = xp*C_w^T.  One block = 32 time rows of one batch, 8 waves.
// A-fragment loads hoisted: one A feeds 2 (stage 1) / 4 (stage 2) WMMAs.
// ---------------------------------------------------------------------------
__global__ __launch_bounds__(256) void fused_front(const float* __restrict__ x,
                                                   const float* __restrict__ cw,
                                                   const float* __restrict__ b_in,
                                                   const _Float16* __restrict__ w16,
                                                   float* __restrict__ xp_g,
                                                   float* __restrict__ bbar_g,
                                                   float* __restrict__ cm_g) {
    __shared__ __align__(16) _Float16 xcL[TROWS * LPAD];
    __shared__ __align__(16) _Float16 xpH[TROWS * LPAD];
    __shared__ float xpF[TROWS * 128];

    int b  = blockIdx.x >> 6;
    int t0 = (blockIdx.x & 63) << 5;
    int tid = threadIdx.x;

    // Stage 0: causal depthwise conv (K=4) + SiLU -> f16 LDS tile [t][d].
    #pragma unroll
    for (int i = 0; i < 16; ++i) {
        int idx = i * 256 + tid;
        int d  = idx >> 5;
        int tl = idx & 31;
        int t  = t0 + tl;
        const float* xr = x + ((size_t)b * DIM + d) * NSEQ;
        float acc = 0.f;
        #pragma unroll
        for (int k = 0; k < 4; ++k) {
            int tg = t - 3 + k;
            if (tg >= 0) acc += xr[tg] * cw[d * 4 + k];
        }
        acc = acc * sigmoidf_fast(acc);           // SiLU
        xcL[tl * LPAD + d] = (_Float16)acc;
    }
    __syncthreads();

    int wv = tid >> 5, lane = tid & 31;
    int m0  = (wv >> 2) << 4;       // {0,16}
    int n0b = (wv & 3) << 5;        // {0,32,64,96}
    int nt0 = n0b >> 4;             // first of the wave's two n-tiles
    int nlo = lane & 15, hf = lane >> 4;
    size_t rowbase = (size_t)b * NSEQ + t0;

    // Stage 1: xp = xc @ W_in^T + b_in
    {
        v8f ac0 = {0.f,0.f,0.f,0.f,0.f,0.f,0.f,0.f};
        v8f ac1 = ac0;
        #pragma unroll
        for (int ck = 0; ck < 4; ++ck) {
            v16h a  = load_a_frag(xcL, m0, ck, lane);
            v16h b0 = load_b_frag(w16, nt0,     ck, lane);
            v16h b1 = load_b_frag(w16, nt0 + 1, ck, lane);
            ac0 = WMMA_F16(a, b0, ac0);
            ac1 = WMMA_F16(a, b1, ac1);
        }
        #pragma unroll
        for (int nt = 0; nt < 2; ++nt) {
            int n = ((nt0 + nt) << 4) + nlo;
            float bi = b_in[n];
            #pragma unroll
            for (int r = 0; r < 8; ++r) {
                int m = m0 + r + 8 * hf;
                float v = (nt == 0 ? ac0[r] : ac1[r]) + bi;
                xpF[m * 128 + n]  = v;
                xpH[m * LPAD + n] = (_Float16)v;
                xp_g[(rowbase + m) * 128 + n] = v;
            }
        }
    }
    __syncthreads();

    // Stage 2: Bbar = (xp @ B_w^T) * sigmoid(xp)  and  Cm = xp @ C_w^T.
    {
        v8f aB0 = {0.f,0.f,0.f,0.f,0.f,0.f,0.f,0.f};
        v8f aB1 = aB0, aC0 = aB0, aC1 = aB0;
        const _Float16* wB = w16 + 1 * 16384;
        const _Float16* wC = w16 + 2 * 16384;
        #pragma unroll
        for (int ck = 0; ck < 4; ++ck) {
            v16h a  = load_a_frag(xpH, m0, ck, lane);
            v16h b0 = load_b_frag(wB, nt0,     ck, lane);
            v16h b1 = load_b_frag(wB, nt0 + 1, ck, lane);
            v16h c0 = load_b_frag(wC, nt0,     ck, lane);
            v16h c1 = load_b_frag(wC, nt0 + 1, ck, lane);
            aB0 = WMMA_F16(a, b0, aB0);
            aB1 = WMMA_F16(a, b1, aB1);
            aC0 = WMMA_F16(a, c0, aC0);
            aC1 = WMMA_F16(a, c1, aC1);
        }
        #pragma unroll
        for (int nt = 0; nt < 2; ++nt) {
            int n = ((nt0 + nt) << 4) + nlo;
            #pragma unroll
            for (int r = 0; r < 8; ++r) {
                int m = m0 + r + 8 * hf;
                float gB = (nt == 0 ? aB0[r] : aB1[r]);
                float gC = (nt == 0 ? aC0[r] : aC1[r]);
                bbar_g[(rowbase + m) * 128 + n] = gB * sigmoidf_fast(xpF[m * 128 + n]);
                cm_g  [(rowbase + m) * 128 + n] = gC;
            }
        }
    }
}

// ---------------------------------------------------------------------------
// Chunked selective scan.  Decay over a chunk factorizes:
//   prod_t exp(A[c,s]*d_t[c]) = exp(A[c,s] * sum_t d_t[c])
// Pass 1: per chunk (zero-init) local final state Hloc + delta-sum S.
// Pass 2: 16-step sequential combine -> chunk entry states H0 (exact).
// Pass 3: re-run each chunk from its true H0, emitting y.  128 blocks live.
// Thread layout (passes 1/3): tid -> (c = tid>>2, s-quarter = tid&3); full
// 128x128 state + A in registers; one barrier per step; shfl_xor dot finish.
// ---------------------------------------------------------------------------
__global__ __launch_bounds__(512) void scan_pass1(const float* __restrict__ bbar,
                                                  const float* __restrict__ xp,
                                                  const float* __restrict__ A_log,
                                                  float* __restrict__ Hloc,
                                                  float* __restrict__ Ssum) {
    __shared__ float buf[2][272];   // [Bbar 0..127 | xp 128..255]

    int b = blockIdx.x >> 4, g = blockIdx.x & (NCH - 1);
    int tid = threadIdx.x;
    int c = tid >> 2, sq = tid & 3, s0 = sq << 5;

    float a[32], h[32];
    const float* alog = A_log + c * DSTATE + s0;
    #pragma unroll
    for (int i = 0; i < 32; ++i) { a[i] = -__expf(alog[i]); h[i] = 0.f; }

    size_t base = ((size_t)b * NSEQ + (size_t)g * CHUNK) * 128;
    int arr = tid >> 7, idx = tid & 127;
    const float* src = (arr == 0) ? bbar : xp;
    if (tid < 256) buf[0][arr * 128 + idx] = src[base + idx];
    __syncthreads();

    float dsum = 0.f;
    for (int t = 0; t < CHUNK; ++t) {
        int cur = t & 1;
        if (tid < 256) {
            if (t + 1 < CHUNK)
                buf[cur ^ 1][arr * 128 + idx] = src[base + (size_t)(t + 1) * 128 + idx];
            if (t + 8 < CHUNK)
                __builtin_prefetch(src + base + (size_t)(t + 8) * 128 + idx, 0, 1);
        }
        float dlt = sigmoidf_fast(buf[cur][128 + c]);
        dsum += dlt;
        const float* bb = &buf[cur][0] + s0;
        #pragma unroll
        for (int i = 0; i < 32; ++i)
            h[i] = __expf(a[i] * dlt) * h[i] + bb[i];
        __syncthreads();
    }

    size_t hbase = (((size_t)(b * NCH + g) * 128 + c) << 7) + s0;
    #pragma unroll
    for (int i = 0; i < 32; ++i) Hloc[hbase + i] = h[i];
    if (sq == 0) Ssum[(b * NCH + g) * 128 + c] = dsum;
}

__global__ __launch_bounds__(256) void scan_pass2(const float* __restrict__ Hloc,
                                                  const float* __restrict__ Ssum,
                                                  const float* __restrict__ A_log,
                                                  float* __restrict__ H0) {
    int gid = blockIdx.x * 256 + threadIdx.x;     // 0 .. 8*128*128-1
    int b = gid >> 14;
    int rem = gid & 16383;
    int c = rem >> 7, s = rem & 127;
    float A = -__expf(A_log[c * DSTATE + s]);
    float h = 0.f;
    #pragma unroll 4
    for (int g = 0; g < NCH; ++g) {
        size_t o = (((size_t)(b * NCH + g) * 128 + c) << 7) + s;
        H0[o] = h;
        h = __expf(A * Ssum[(b * NCH + g) * 128 + c]) * h + Hloc[o];
    }
}

__global__ __launch_bounds__(512) void scan_pass3(const float* __restrict__ bbar,
                                                  const float* __restrict__ cm,
                                                  const float* __restrict__ xp,
                                                  const float* __restrict__ A_log,
                                                  const float* __restrict__ Dp,
                                                  const float* __restrict__ H0,
                                                  float* __restrict__ y) {
    __shared__ float buf[2][400];   // [Bbar | Cm | xp]

    int b = blockIdx.x >> 4, g = blockIdx.x & (NCH - 1);
    int tid = threadIdx.x;
    int c = tid >> 2, sq = tid & 3, s0 = sq << 5;

    float a[32], h[32];
    const float* alog = A_log + c * DSTATE + s0;
    size_t hbase = (((size_t)(b * NCH + g) * 128 + c) << 7) + s0;
    #pragma unroll
    for (int i = 0; i < 32; ++i) { a[i] = -__expf(alog[i]); h[i] = H0[hbase + i]; }
    float Dc = Dp[c];

    size_t base = ((size_t)b * NSEQ + (size_t)g * CHUNK) * 128;
    int arr = tid >> 7, idx = tid & 127;
    const float* src = (arr == 0) ? bbar : (arr == 1) ? cm : xp;
    if (tid < 384) buf[0][arr * 128 + idx] = src[base + idx];
    __syncthreads();

    for (int t = 0; t < CHUNK; ++t) {
        int cur = t & 1;
        if (tid < 384) {
            if (t + 1 < CHUNK)
                buf[cur ^ 1][arr * 128 + idx] = src[base + (size_t)(t + 1) * 128 + idx];
            if (t + 8 < CHUNK)
                __builtin_prefetch(src + base + (size_t)(t + 8) * 128 + idx, 0, 1);
        }
        float xpc = buf[cur][256 + c];
        float dlt = sigmoidf_fast(xpc);
        const float* bb = &buf[cur][0]   + s0;
        const float* cc = &buf[cur][128] + s0;
        float acc = 0.f;
        #pragma unroll
        for (int i = 0; i < 32; ++i) {
            h[i] = __expf(a[i] * dlt) * h[i] + bb[i];
            acc += h[i] * cc[i];
        }
        acc += __shfl_xor(acc, 1);
        acc += __shfl_xor(acc, 2);
        if (sq == 0) y[base + (size_t)t * 128 + c] = acc + Dc * xpc;
        __syncthreads();
    }
}

// ---------------------------------------------------------------------------
// Back end: out = y @ W_out^T + b_out + x^T, LayerNorm over dim, transposed
// coalesced store to [B, DIM, N].
// ---------------------------------------------------------------------------
__global__ __launch_bounds__(256) void fused_back(const float* __restrict__ y_g,
                                                  const float* __restrict__ x,
                                                  const float* __restrict__ b_out,
                                                  const float* __restrict__ gamma,
                                                  const float* __restrict__ beta,
                                                  const _Float16* __restrict__ w16,
                                                  float* __restrict__ out) {
    __shared__ __align__(16) _Float16 yL[TROWS * LPAD];
    __shared__ float oT[TROWS * 132];

    int b  = blockIdx.x >> 6;
    int t0 = (blockIdx.x & 63) << 5;
    int tid = threadIdx.x;

    #pragma unroll
    for (int i = 0; i < 16; ++i) {
        int idx = i * 256 + tid;
        int tl = idx >> 7, cc = idx & 127;
        yL[tl * LPAD + cc] = (_Float16)y_g[((size_t)b * NSEQ + t0 + tl) * 128 + cc];
    }
    __syncthreads();

    int wv = tid >> 5, lane = tid & 31;
    int m0  = (wv >> 2) << 4;
    int n0b = (wv & 3) << 5;
    int nt0 = n0b >> 4;
    int nlo = lane & 15, hf = lane >> 4;

    {
        v8f ac0 = {0.f,0.f,0.f,0.f,0.f,0.f,0.f,0.f};
        v8f ac1 = ac0;
        const _Float16* wO = w16 + 3 * 16384;
        #pragma unroll
        for (int ck = 0; ck < 4; ++ck) {
            v16h a  = load_a_frag(yL, m0, ck, lane);
            v16h b0 = load_b_frag(wO, nt0,     ck, lane);
            v16h b1 = load_b_frag(wO, nt0 + 1, ck, lane);
            ac0 = WMMA_F16(a, b0, ac0);
            ac1 = WMMA_F16(a, b1, ac1);
        }
        #pragma unroll
        for (int nt = 0; nt < 2; ++nt) {
            int n = ((nt0 + nt) << 4) + nlo;
            float bo = b_out[n];
            const float* xr = x + ((size_t)b * DIM + n) * NSEQ + t0;
            #pragma unroll
            for (int r = 0; r < 8; ++r) {
                int m = m0 + r + 8 * hf;
                oT[m * 132 + n] = (nt == 0 ? ac0[r] : ac1[r]) + bo + xr[m];
            }
        }
    }
    __syncthreads();

    // LayerNorm over dim=128: 4 rows per wave, 4 elements per lane.
    #pragma unroll
    for (int j = 0; j < 4; ++j) {
        int row = wv * 4 + j;
        float v[4], s = 0.f, s2 = 0.f;
        #pragma unroll
        for (int q = 0; q < 4; ++q) {
            v[q] = oT[row * 132 + lane * 4 + q];
            s += v[q]; s2 += v[q] * v[q];
        }
        #pragma unroll
        for (int m = 1; m < 32; m <<= 1) {
            s  += __shfl_xor(s,  m);
            s2 += __shfl_xor(s2, m);
        }
        float mu  = s * (1.f / 128.f);
        float var = s2 * (1.f / 128.f) - mu * mu;
        float rs  = rsqrtf(var + 1e-5f);
        #pragma unroll
        for (int q = 0; q < 4; ++q) {
            int cidx = lane * 4 + q;
            oT[row * 132 + cidx] = (v[q] - mu) * rs * gamma[cidx] + beta[cidx];
        }
    }
    __syncthreads();

    #pragma unroll
    for (int i = 0; i < 16; ++i) {
        int idx = i * 256 + tid;
        int cidx = idx >> 5, tl = idx & 31;
        out[((size_t)b * DIM + cidx) * NSEQ + t0 + tl] = oT[tl * 132 + cidx];
    }
}

// ---------------------------------------------------------------------------
extern "C" void kernel_launch(void* const* d_in, const int* in_sizes, int n_in,
                              void* d_out, int out_size, void* d_ws, size_t ws_size,
                              hipStream_t stream) {
    const float* x       = (const float*)d_in[0];
    const float* conv_w  = (const float*)d_in[1];
    const float* A_log   = (const float*)d_in[2];
    const float* D_param = (const float*)d_in[3];
    const float* B_w     = (const float*)d_in[4];
    const float* C_w     = (const float*)d_in[5];
    const float* W_in    = (const float*)d_in[6];
    const float* b_in    = (const float*)d_in[7];
    const float* W_out   = (const float*)d_in[8];
    const float* b_out   = (const float*)d_in[9];
    const float* gamma   = (const float*)d_in[10];
    const float* beta    = (const float*)d_in[11];

    char* ws = (char*)d_ws;
    const size_t MAT = (size_t)BSZ * NSEQ * 128 * sizeof(float);       // 8 MB
    const size_t HMT = (size_t)BSZ * NCH * 128 * 128 * sizeof(float);  // 8 MB
    _Float16* w16 = (_Float16*)ws;
    size_t off = 4 * 128 * 128 * sizeof(_Float16);
    off = (off + 255) & ~(size_t)255;
    float* xp_g  = (float*)(ws + off); off += MAT;
    float* bbar  = (float*)(ws + off); off += MAT;
    float* cm    = (float*)(ws + off); off += MAT;
    float* y_g   = (float*)(ws + off); off += MAT;
    float* Hloc  = (float*)(ws + off); off += HMT;
    float* H0    = (float*)(ws + off); off += HMT;
    float* Ssum  = (float*)(ws + off); off += (size_t)BSZ * NCH * 128 * sizeof(float);

    prep_weights<<<16, 256, 0, stream>>>(W_in, B_w, C_w, W_out, w16);
    fused_front<<<BSZ * (NSEQ / TROWS), 256, 0, stream>>>(x, conv_w, b_in, w16,
                                                          xp_g, bbar, cm);
    scan_pass1<<<BSZ * NCH, 512, 0, stream>>>(bbar, xp_g, A_log, Hloc, Ssum);
    scan_pass2<<<(BSZ * 128 * 128) / 256, 256, 0, stream>>>(Hloc, Ssum, A_log, H0);
    scan_pass3<<<BSZ * NCH, 512, 0, stream>>>(bbar, cm, xp_g, A_log, D_param, H0, y_g);
    fused_back<<<BSZ * (NSEQ / TROWS), 256, 0, stream>>>(y_g, x, b_out, gamma, beta,
                                                         w16, (float*)d_out);
}
